// ProjectedVQ_26104811225631
// MI455X (gfx1250) — matrix operand order repaired and checked by reading, not compiled, not run on hardware
//
#include <hip/hip_runtime.h>
#include <cstdint>
#include <cstddef>

typedef __attribute__((ext_vector_type(16))) __bf16 v16bf;
typedef __attribute__((ext_vector_type(8)))  float  v8f;
typedef __attribute__((ext_vector_type(2)))  float  v2f;
typedef __attribute__((ext_vector_type(4)))  int    v4i;
typedef unsigned short u16;
typedef unsigned char  u8;

#define E_DIM 16384
#define D_DIM 768      // also Q = B*R = 768, also K of both big GEMMs
#define NB    32
#define NC    256
#define NR    24

union Frag16 { uint4 q[2]; v16bf v; };

__device__ inline u16 f2bf(float f) {
  unsigned u = __float_as_uint(f);
  u = u + 0x7FFFu + ((u >> 16) & 1u);       // round-to-nearest-even
  return (u16)(u >> 16);
}

__device__ inline v8f wmma_bf16(v16bf a, v16bf b, v8f c) {
  return __builtin_amdgcn_wmma_f32_16x16x32_bf16(false, a, false, b, (short)0, c, false, false);
}
__device__ inline v8f wmma_f32x4(v2f a, v2f b, v8f c) {
  return __builtin_amdgcn_wmma_f32_16x16x4_f32(false, a, false, b, (short)0, c, false, false);
}

// ---- CDNA5 async global->LDS copy (ASYNCcnt-tracked), builtin if present else inline asm ----
__device__ inline void async_b128(const u16* g, u16* l) {
#if __has_builtin(__builtin_amdgcn_global_load_async_to_lds_b128)
  __builtin_amdgcn_global_load_async_to_lds_b128(
      (__attribute__((address_space(1))) v4i*)g,       // global src (v4i global*, const dropped)
      (__attribute__((address_space(3))) v4i*)l,       // LDS dst
      0, 0);
#else
  unsigned lo = (unsigned)(uintptr_t)(void*)l;   // low 32 bits of generic ptr = LDS byte offset
  asm volatile("global_load_async_to_lds_b128 %0, %1, off" :: "v"(lo), "v"(g) : "memory");
#endif
}

__device__ inline void wait_async0() {
#if __has_builtin(__builtin_amdgcn_s_wait_asynccnt)
  __builtin_amdgcn_s_wait_asynccnt(0);
#else
  asm volatile("s_wait_asynccnt 0x0" ::: "memory");
#endif
}

// ---------------- prep kernels ----------------

__global__ void k_convert_x(const float* __restrict__ x, u16* __restrict__ xb, int n) {
  int i = blockIdx.x * 256 + threadIdx.x;
  if (i < n) xb[i] = f2bf(x[i]);
}

__global__ void k_make_P(const float* __restrict__ proj, const u8* __restrict__ bdead,
                         u16* __restrict__ Pb, u16* __restrict__ Pt) {
  int i = blockIdx.x * 256 + threadIdx.x;          // < 768*768
  int q = i / D_DIM, d = i - q * D_DIM;
  float v = proj[i];
  if (bdead[q / NR]) v = 0.f;
  u16 h = f2bf(v);
  Pb[i] = h;                                       // [q][d]  (B-tiles of GEMM1)
  Pt[(size_t)d * D_DIM + q] = h;                   // [d][q]  (B-tiles of GEMM2)
}

__global__ void k_make_cb(const float* __restrict__ cb, const u8* __restrict__ cdead,
                          u16* __restrict__ cbT, float* __restrict__ cnormp) {
  int i = blockIdx.x * 256 + threadIdx.x;          // < 32*256,  b = i>>8, c = i&255
  int b = i >> 8, c = i & 255;
  float s = 0.f;
  for (int r = 0; r < NR; ++r) {
    float v = cb[(size_t)i * NR + r];
    s += v * v;
    cbT[((size_t)(b * 32 + r)) * NC + c] = f2bf(v);   // [b][r(pad32)][c]
  }
  for (int r = NR; r < 32; ++r) cbT[((size_t)(b * 32 + r)) * NC + c] = 0;
  cnormp[i] = s + (cdead[i] ? 1e20f : 0.f);
}

__global__ void k_zb(const float* __restrict__ proj, const float* __restrict__ biases,
                     const u8* __restrict__ bdead, float* __restrict__ zb) {
  int q = blockIdx.x, t = threadIdx.x, b = q / NR;
  float s = 0.f;
  for (int d = t; d < D_DIM; d += 64) s += biases[b * D_DIM + d] * proj[(size_t)q * D_DIM + d];
  __shared__ float red[64];
  red[t] = s; __syncthreads();
  for (int o = 32; o > 0; o >>= 1) { if (t < o) red[t] += red[t + o]; __syncthreads(); }
  if (t == 0) zb[q] = bdead[b] ? 0.f : red[0];
}

__global__ void k_bias_sum(const float* __restrict__ biases, const u8* __restrict__ bdead,
                           float* __restrict__ bsum) {
  int d = blockIdx.x * 256 + threadIdx.x;
  if (d >= D_DIM) return;
  float s = 0.f;
  for (int b = 0; b < NB; ++b) if (!bdead[b]) s += biases[b * D_DIM + d];
  bsum[d] = s;
}

__global__ void k_rnorm(const float* __restrict__ proj, float* __restrict__ invn) {
  int q = blockIdx.x, t = threadIdx.x;
  float s = 0.f;
  for (int d = t; d < D_DIM; d += 64) { float v = proj[(size_t)q * D_DIM + d]; s += v * v; }
  __shared__ float red[64];
  red[t] = s; __syncthreads();
  for (int o = 32; o > 0; o >>= 1) { if (t < o) red[t] += red[t + o]; __syncthreads(); }
  if (t == 0) invn[q] = 1.f / fmaxf(sqrtf(red[0]), 1e-12f);
}

__global__ void k_orth(const float* __restrict__ proj, const float* __restrict__ invn,
                       float* __restrict__ orthp) {
  int pair = blockIdx.x, t = threadIdx.x;
  int bb = pair >> 5, cc = pair & 31;
  float s = 0.f;
  for (int j = t; j < NR * D_DIM; j += 256) {
    int r = j / D_DIM;
    s += proj[(size_t)bb * (NR * D_DIM) + j] * proj[(size_t)cc * (NR * D_DIM) + j]
       * invn[bb * NR + r] * invn[cc * NR + r];
  }
  __shared__ float red[256];
  red[t] = s; __syncthreads();
  for (int o = 128; o > 0; o >>= 1) { if (t < o) red[t] += red[t + o]; __syncthreads(); }
  if (t == 0) { float d = red[0] - (bb == cc ? 1.f : 0.f); orthp[pair] = d * d; }
}

// ---------------- big GEMMs (bf16 WMMA, async double-buffered LDS) ----------------
// C(E x 768) = A(E x 768, bf16) * BT^T, BT stored as [n][k] so B-tiles load row-contiguous.
// MODE 0: out = acc - colvec[n]            (z = xP - zb)
// MODE 1: rec = acc + colvec[n]; out=rec; per-block Σ(rec-x)^2 -> msep  (recon + mse)
template <int MODE>
__global__ __launch_bounds__(256)
void k_gemm(const u16* __restrict__ A, const u16* __restrict__ BT,
            float* __restrict__ outp, const float* __restrict__ colvec,
            const float* __restrict__ xref, float* __restrict__ msep)
{
  __shared__ u16 sA[2][128 * 40];  // 128 rows x 32 k, stride 40 (16B aligned, conflict-free)
  __shared__ u16 sB[2][128 * 40];
  const int tid  = threadIdx.x;
  const int wave = tid >> 5, lane = tid & 31;
  const int wm = wave & 3, wn = wave >> 2;        // 4 x 2 wave grid -> 32 x 64 per wave
  const int lm = lane & 15, hi = lane >> 4;
  const int tm = blockIdx.x, tn = blockIdx.y;

  const v8f zero = {0.f,0.f,0.f,0.f,0.f,0.f,0.f,0.f};
  v8f acc[2][4];
  #pragma unroll
  for (int mi = 0; mi < 2; ++mi)
    #pragma unroll
    for (int ni = 0; ni < 4; ++ni) acc[mi][ni] = zero;

  const int row = tid >> 1, half = tid & 1;       // 2 threads/row, 16 u16 (32B) each
  const u16* gA = A  + (size_t)(tm * 128 + row) * D_DIM + half * 16;
  const u16* gB = BT + (size_t)(tn * 128 + row) * D_DIM + half * 16;
  const int lofs = row * 40 + half * 16;

  // prologue: stage 0 -> buffer 0 (ASYNCcnt-tracked DMA into LDS)
  async_b128(gA,     &sA[0][lofs]);
  async_b128(gA + 8, &sA[0][lofs + 8]);
  async_b128(gB,     &sB[0][lofs]);
  async_b128(gB + 8, &sB[0][lofs + 8]);

  for (int it = 0; it < D_DIM / 32; ++it) {
    const int cur = it & 1, nxt = cur ^ 1;
    wait_async0();                 // stage `it` landed (issued one compute-phase ago)
    __syncthreads();               // all waves arrived; all reads of buf `nxt` finished
    if (it + 1 < D_DIM / 32) {     // prefetch stage it+1 into the other buffer
      const int k0 = (it + 1) * 32;
      async_b128(gA + k0,     &sA[nxt][lofs]);
      async_b128(gA + k0 + 8, &sA[nxt][lofs + 8]);
      async_b128(gB + k0,     &sB[nxt][lofs]);
      async_b128(gB + k0 + 8, &sB[nxt][lofs + 8]);
    }

    Frag16 fa[2], fb[4];
    #pragma unroll
    for (int mi = 0; mi < 2; ++mi) {              // A frag: lane<16: K0-15 ; lane>=16: K16-31
      const u16* p = &sA[cur][(wm * 32 + mi * 16 + lm) * 40 + hi * 16];
      fa[mi].q[0] = *(const uint4*)p;
      fa[mi].q[1] = *(const uint4*)(p + 8);
    }
    #pragma unroll
    for (int ni = 0; ni < 4; ++ni) {              // B frag: lane<16: K0-15 ; lane>=16: K16-31
      const u16* p = &sB[cur][(wn * 64 + ni * 16 + lm) * 40 + hi * 16];
      fb[ni].q[0] = *(const uint4*)p;
      fb[ni].q[1] = *(const uint4*)(p + 8);
    }
    #pragma unroll
    for (int mi = 0; mi < 2; ++mi)
      #pragma unroll
      for (int ni = 0; ni < 4; ++ni)
        acc[mi][ni] = wmma_bf16(fa[mi].v, fb[ni].v, acc[mi][ni]);
  }

  float local = 0.f;
  #pragma unroll
  for (int mi = 0; mi < 2; ++mi)
    #pragma unroll
    for (int ni = 0; ni < 4; ++ni) {
      int col = tn * 128 + wn * 64 + ni * 16 + lm;
      float cv = colvec[col];
      #pragma unroll
      for (int vv = 0; vv < 8; ++vv) {
        int rowg = tm * 128 + wm * 32 + mi * 16 + vv + 8 * hi;
        size_t idx = (size_t)rowg * D_DIM + col;
        float val = acc[mi][ni][vv];
        if (MODE == 0) {
          outp[idx] = val - cv;
        } else {
          float rec = val + cv;
          outp[idx] = rec;
          float dd = rec - xref[idx];
          local += dd * dd;
        }
      }
    }
  if (MODE == 1) {
    #pragma unroll
    for (int m = 16; m >= 1; m >>= 1) local += __shfl_xor(local, m);
    __shared__ float wred[8];
    if (lane == 0) wred[wave] = local;
    __syncthreads();
    if (tid == 0) {
      float s = 0.f;
      for (int i = 0; i < 8; ++i) s += wred[i];
      msep[blockIdx.x * gridDim.y + blockIdx.y] = s;
    }
  }
}

// ---------------- fused VQ kernel: dist2 -> softmax -> entropy -> recon_local ----------------
// one wave handles 16 e-rows of one book
__global__ __launch_bounds__(32)
void k_vq(const float* __restrict__ zbuf, const float* __restrict__ cb,
          const float* __restrict__ cnormp, const float* __restrict__ kvec,
          const u16* __restrict__ cbT, u16* __restrict__ rl, float* __restrict__ entp)
{
  const int et = blockIdx.x, b = blockIdx.y;
  const int lane = threadIdx.x;
  const int lm = lane & 15, hi = lane >> 4;
  const int e0 = et * 16;
  __shared__ u16 sp[16 * 256];                    // prob tile, bf16

  const v8f zero = {0.f,0.f,0.f,0.f,0.f,0.f,0.f,0.f};
  v8f acc[16];
  #pragma unroll
  for (int ct = 0; ct < 16; ++ct) acc[ct] = zero;

  // z . c^T via V_WMMA_F32_16X16X4_F32, K = 24 = 6 x 4
  const int kb2 = hi ? 2 : 0;
  const float* zr = zbuf + (size_t)(e0 + lm) * D_DIM + b * NR + kb2;
  #pragma unroll
  for (int kk = 0; kk < 6; ++kk) {
    v2f av; av.x = zr[kk * 4]; av.y = zr[kk * 4 + 1];
    #pragma unroll
    for (int ct = 0; ct < 16; ++ct) {
      const float* cp = cb + ((size_t)(b * NC + ct * 16 + lm)) * NR + kk * 4 + kb2;
      v2f bv; bv.x = cp[0]; bv.y = cp[1];
      acc[ct] = wmma_f32x4(av, bv, acc[ct]);
    }
  }

  // logits: row-constant |z|^2 drops out of softmax -> logit = k*(2 z.c - |c|^2 - dead*1e20)
  float kp = fmaxf(kvec[b], 1e-9f);
  float cn[16];
  #pragma unroll
  for (int ct = 0; ct < 16; ++ct) cn[ct] = cnormp[b * NC + ct * 16 + lm];
  #pragma unroll
  for (int ct = 0; ct < 16; ++ct)
    #pragma unroll
    for (int vv = 0; vv < 8; ++vv)
      acc[ct][vv] = kp * (2.0f * acc[ct][vv] - cn[ct]);

  // softmax over 256 codes: row m = vv + 8*hi lives in one VGPR across one 16-lane half
  float rmv[8], smv[8];
  #pragma unroll
  for (int vv = 0; vv < 8; ++vv) {
    float m = -3.4e38f;
    #pragma unroll
    for (int ct = 0; ct < 16; ++ct) m = fmaxf(m, acc[ct][vv]);
    m = fmaxf(m, __shfl_xor(m, 1));
    m = fmaxf(m, __shfl_xor(m, 2));
    m = fmaxf(m, __shfl_xor(m, 4));
    m = fmaxf(m, __shfl_xor(m, 8));
    rmv[vv] = m;
    float s = 0.f;
    #pragma unroll
    for (int ct = 0; ct < 16; ++ct) s += __expf(acc[ct][vv] - m);
    s += __shfl_xor(s, 1);
    s += __shfl_xor(s, 2);
    s += __shfl_xor(s, 4);
    s += __shfl_xor(s, 8);
    smv[vv] = s;
  }

  // prob -> LDS (bf16) + entropy
  float enth = 0.f;
  #pragma unroll
  for (int vv = 0; vv < 8; ++vv) {
    float lsum = __logf(smv[vv]);
    float invs = 1.f / smv[vv];
    float ev = 0.f;
    #pragma unroll
    for (int ct = 0; ct < 16; ++ct) {
      float t = acc[ct][vv] - rmv[vv];
      float p = __expf(t) * invs;
      ev -= p * (t - lsum);
      sp[(vv + 8 * hi) * 256 + ct * 16 + lm] = f2bf(p);
    }
    ev += __shfl_xor(ev, 1);
    ev += __shfl_xor(ev, 2);
    ev += __shfl_xor(ev, 4);
    ev += __shfl_xor(ev, 8);
    enth += ev;
  }
  float etot = enth + __shfl_xor(enth, 16);

  __syncthreads();

  // recon_local = prob (16x256) * cbT (256 x 32pad), bf16 WMMA, 8 k-steps x 2 n-tiles
  v8f r0 = zero, r1 = zero;
  #pragma unroll
  for (int ks = 0; ks < 8; ++ks) {
    Frag16 fa;
    const u16* ap = &sp[lm * 256 + ks * 32 + hi * 16];
    fa.q[0] = *(const uint4*)ap;
    fa.q[1] = *(const uint4*)(ap + 8);
    Frag16 fb0, fb1;
    const u16* bp0 = cbT + ((size_t)(b * 32 + lm)) * NC + ks * 32 + hi * 16;
    const u16* bp1 = cbT + ((size_t)(b * 32 + 16 + lm)) * NC + ks * 32 + hi * 16;
    fb0.q[0] = *(const uint4*)bp0; fb0.q[1] = *(const uint4*)(bp0 + 8);
    fb1.q[0] = *(const uint4*)bp1; fb1.q[1] = *(const uint4*)(bp1 + 8);
    r0 = wmma_bf16(fa.v, fb0.v, r0);
    r1 = wmma_bf16(fa.v, fb1.v, r1);
  }

  if (lane == 0) entp[b * 1024 + et] = etot;

  #pragma unroll
  for (int vv = 0; vv < 8; ++vv) {
    int e = e0 + vv + 8 * hi;
    rl[(size_t)e * D_DIM + b * NR + lm] = f2bf(r0[vv]);
    if (lm < 8) rl[(size_t)e * D_DIM + b * NR + 16 + lm] = f2bf(r1[vv]);
  }
}

// ---------------- scalar finalize (deterministic, fixed-order sums) ----------------
__global__ void k_finalize(const float* __restrict__ msep, const float* __restrict__ entp,
                           const float* __restrict__ orthp, const u8* __restrict__ bdead,
                           float* __restrict__ out)
{
  if (threadIdx.x != 0) return;
  float ms = 0.f;
  for (int i = 0; i < 768; ++i) ms += msep[i];
  float orth = 0.f;
  for (int i = 0; i < 1024; ++i) orth += orthp[i];
  int nact = 0;
  for (int b = 0; b < NB; ++b) nact += bdead[b] ? 0 : 1;
  if (nact < 1) nact = 1;
  float ent = 0.f;
  for (int b = 0; b < NB; ++b) {
    if (bdead[b]) continue;
    float s = 0.f;
    for (int t = 0; t < 1024; ++t) s += entp[b * 1024 + t];
    ent += s * (1.0f / (float)E_DIM);
  }
  float mse = ms / ((float)E_DIM * (float)D_DIM);
  out[0] = 1.25f * mse - 0.03f * (ent / (float)nact) + orth * (1.0f / 1024.0f);
}

// ---------------- host launcher ----------------
extern "C" void kernel_launch(void* const* d_in, const int* in_sizes, int n_in,
                              void* d_out, int out_size, void* d_ws, size_t ws_size,
                              hipStream_t stream) {
  (void)in_sizes; (void)n_in; (void)out_size; (void)ws_size;
  const float* x      = (const float*)d_in[0];
  const float* proj   = (const float*)d_in[1];
  const float* cb     = (const float*)d_in[2];
  const float* kvec   = (const float*)d_in[3];
  const float* biases = (const float*)d_in[4];
  const u8*    cdead  = (const u8*)d_in[5];
  const u8*    bdead  = (const u8*)d_in[6];

  u8* base = (u8*)d_ws;
  size_t off = 0;
  auto take = [&](size_t n) -> void* {
    void* p = base + off;
    off = (off + n + 255) & ~(size_t)255;
    return p;
  };
  u16*   xb     = (u16*)  take((size_t)E_DIM * D_DIM * 2);   // x in bf16
  u16*   Pb     = (u16*)  take((size_t)D_DIM * D_DIM * 2);   // P [q][d]
  u16*   Pt     = (u16*)  take((size_t)D_DIM * D_DIM * 2);   // P [d][q]
  float* zbuf   = (float*)take((size_t)E_DIM * D_DIM * 4);   // z
  u16*   rl     = (u16*)  take((size_t)E_DIM * D_DIM * 2);   // recon_local bf16
  u16*   cbT    = (u16*)  take((size_t)NB * 32 * NC * 2);    // codebook [b][r pad32][c]
  float* cnormp = (float*)take((size_t)NB * NC * 4);
  float* zb     = (float*)take((size_t)D_DIM * 4);
  float* bsum   = (float*)take((size_t)D_DIM * 4);
  float* invn   = (float*)take((size_t)D_DIM * 4);
  float* msep   = (float*)take(768 * 4);
  float* entp   = (float*)take((size_t)NB * 1024 * 4);
  float* orthp  = (float*)take(1024 * 4);

  float* loss_out  = (float*)d_out;
  float* recon_out = (float*)d_out + 1;

  // prep
  k_convert_x<<<dim3((E_DIM * D_DIM) / 256), dim3(256), 0, stream>>>(x, xb, E_DIM * D_DIM);
  k_make_P   <<<dim3((D_DIM * D_DIM) / 256), dim3(256), 0, stream>>>(proj, bdead, Pb, Pt);
  k_make_cb  <<<dim3((NB * NC) / 256),       dim3(256), 0, stream>>>(cb, cdead, cbT, cnormp);
  k_zb       <<<dim3(D_DIM), dim3(64),  0, stream>>>(proj, biases, bdead, zb);
  k_bias_sum <<<dim3(3),     dim3(256), 0, stream>>>(biases, bdead, bsum);
  k_rnorm    <<<dim3(D_DIM), dim3(64),  0, stream>>>(proj, invn);
  k_orth     <<<dim3(NB * NB), dim3(256), 0, stream>>>(proj, invn, orthp);

  // z = x P^T - zb
  k_gemm<0><<<dim3(E_DIM / 128, D_DIM / 128), dim3(256), 0, stream>>>(
      xb, Pb, zbuf, zb, nullptr, nullptr);

  // fused VQ: dist2/softmax/entropy/recon_local
  k_vq<<<dim3(E_DIM / 16, NB), dim3(32), 0, stream>>>(zbuf, cb, cnormp, kvec, cbT, rl, entp);

  // recon = rl P + bias_sum, with fused mse partials
  k_gemm<1><<<dim3(E_DIM / 128, D_DIM / 128), dim3(256), 0, stream>>>(
      rl, Pt, recon_out, bsum, x, msep);

  // loss
  k_finalize<<<dim3(1), dim3(32), 0, stream>>>(msep, entp, orthp, bdead, loss_out);
}
